// AttentionBlock_55972013802140
// MI455X (gfx1250) — matrix operand order repaired and checked by reading, not compiled, not run on hardware
//
#include <hip/hip_runtime.h>

typedef __attribute__((ext_vector_type(16))) _Float16 v16h;
typedef __attribute__((ext_vector_type(8)))  _Float16 v8h;
typedef __attribute__((ext_vector_type(8)))  float    v8f;
typedef int v4i __attribute__((vector_size(16)));
typedef __attribute__((address_space(1))) v4i* gv4i_ptr;   // global (prints __device__)
typedef __attribute__((address_space(3))) v4i* lv4i_ptr;   // LDS    (prints __shared__)

#define BATCH 8
#define CH    512
#define NSP   1024          // H*W
#define NHEADS 8
#define HD    64
#define NGRP  32
#define CPG   (CH / NGRP)   // 16
#define QKV_O (3 * CH)      // 1536
#define LDB   520           // padded LDS row stride (halves): bank-conflict-free
#define PLD   40            // padded P-tile row stride (halves)

#if __has_builtin(__builtin_amdgcn_global_load_async_to_lds_b128)
#define ASYNC_LDS 1
#endif

__device__ __forceinline__ int lane_id() { return (int)(threadIdx.x & 31); }

__device__ __forceinline__ v8f zero_v8f() {
  v8f z;
#pragma unroll
  for (int i = 0; i < 8; ++i) z[i] = 0.0f;
  return z;
}

__device__ __forceinline__ v8f wmma_f16(v16h a, v16h b, v8f c) {
  return __builtin_amdgcn_wmma_f32_16x16x32_f16(false, a, false, b, (short)0, c,
                                                false, false);
}

// A fragment (16x32 f16) from row-major [M,K]:
// lane holds row (row_base + lane&15); k(e) = 16*hv + e -> contiguous 16 halves
__device__ __forceinline__ v16h load_a_frag(const _Float16* base, int ld,
                                            int row_base, int k_base) {
  const int l = lane_id();
  const int r = l & 15, hv = l >> 4;
  const _Float16* p = base + (size_t)(row_base + r) * ld + k_base + hv * 16;
  union { v16h v; v8h h[2]; } u;
  u.h[0] = *(const v8h*)(p);
  u.h[1] = *(const v8h*)(p + 8);
  return u.v;
}

// B fragment (32x16 f16) from transposed storage T[col][k] row-major:
// lane holds col (col_base + lane&15); k(e) = 16*hv + e -> contiguous 16 halves
__device__ __forceinline__ v16h load_b_frag(const _Float16* base, int ld,
                                            int col_base, int k_base) {
  const int l = lane_id();
  const int c = l & 15, hv = l >> 4;
  const _Float16* p = base + (size_t)(col_base + c) * ld + k_base + hv * 16;
  union { v16h v; v8h h[2]; } u;
  u.h[0] = *(const v8h*)(p);
  u.h[1] = *(const v8h*)(p + 8);
  return u.v;
}

// Stage a 64 x CH f16 panel (rows contiguous at src, stride CH) into LDS with
// padded stride LDB. 128 threads. Uses async global->LDS copies when available.
__device__ __forceinline__ void stage_panel64(const _Float16* src,
                                              _Float16* dst_lds) {
  const int t = (int)threadIdx.x;            // 128 threads
#pragma unroll
  for (int g = t; g < 64 * (CH / 8); g += 128) {
    const int r = g >> 6;
    const int col = (g & 63) * 8;
    const _Float16* s = src + (size_t)r * CH + col;
    _Float16* d = dst_lds + r * LDB + col;
#if defined(ASYNC_LDS)
    __builtin_amdgcn_global_load_async_to_lds_b128((gv4i_ptr)s, (lv4i_ptr)d, 0, 0);
#else
    *(v8h*)d = *(const v8h*)s;
#endif
  }
#if defined(ASYNC_LDS)
#if __has_builtin(__builtin_amdgcn_s_wait_asynccnt)
  __builtin_amdgcn_s_wait_asynccnt(0);
#else
  asm volatile("s_wait_asynccnt 0" ::: "memory");
#endif
#endif
  __syncthreads();
}

// ---------------- weights fp32 -> f16 ----------------
__global__ void __launch_bounds__(256)
convert_weights_kernel(const float* __restrict__ qkvw,
                       const float* __restrict__ projw,
                       _Float16* __restrict__ qkvw_h,
                       _Float16* __restrict__ projw_h) {
  const int n_qkv = QKV_O * CH;
  const int total = n_qkv + CH * CH;
  int i = blockIdx.x * 256 + threadIdx.x;
  if (i < n_qkv) qkvw_h[i] = (_Float16)qkvw[i];
  else if (i < total) projw_h[i - n_qkv] = (_Float16)projw[i - n_qkv];
}

// ---------------- GroupNorm -> xnT [B,N,C] f16 ----------------
__global__ void __launch_bounds__(256)
groupnorm_kernel(const float* __restrict__ x, const float* __restrict__ gw,
                 const float* __restrict__ gb, _Float16* __restrict__ xnT) {
  const int b = blockIdx.x / NGRP;
  const int g = blockIdx.x % NGRP;
  const int tid = threadIdx.x;
  const float* xg = x + ((size_t)b * CH + g * CPG) * NSP;  // [CPG][NSP]
  float s = 0.f, s2 = 0.f;
  for (int i = tid; i < CPG * NSP; i += 256) {
    float v = xg[i];
    s += v; s2 += v * v;
  }
#pragma unroll
  for (int m = 16; m >= 1; m >>= 1) {
    s  += __shfl_xor(s,  m, 32);
    s2 += __shfl_xor(s2, m, 32);
  }
  __shared__ float red[2][8];
  const int w = tid >> 5;
  if ((tid & 31) == 0) { red[0][w] = s; red[1][w] = s2; }
  __syncthreads();
  if (tid == 0) {
    float a = 0.f, a2 = 0.f;
#pragma unroll
    for (int i = 0; i < 8; ++i) { a += red[0][i]; a2 += red[1][i]; }
    red[0][0] = a; red[1][0] = a2;
  }
  __syncthreads();
  const float inv_n = 1.0f / (float)(CPG * NSP);
  const float mean = red[0][0] * inv_n;
  const float var  = red[1][0] * inv_n - mean * mean;
  const float rstd = rsqrtf(var + 1e-5f);
  for (int i = tid; i < CPG * NSP; i += 256) {
    const int cl = i >> 10;        // i / NSP
    const int n  = i & (NSP - 1);
    const int c  = g * CPG + cl;
    const float v = (xg[i] - mean) * rstd * gw[c] + gb[c];
    xnT[((size_t)b * NSP + n) * CH + c] = (_Float16)v;
  }
}

// ------- QKV GEMM (LDS-staged B panel): qT,kT [B,H,N,hd], v [B,H,hd,N] -------
__global__ void __launch_bounds__(128)
qkv_gemm_kernel(const _Float16* __restrict__ w, const float* __restrict__ bias,
                const _Float16* __restrict__ xnT,
                _Float16* __restrict__ qT, _Float16* __restrict__ kT,
                _Float16* __restrict__ vbuf) {
  __shared__ _Float16 Bs[64 * LDB];
  const int n0 = blockIdx.y * 64;
  const int b = blockIdx.z;
  stage_panel64(xnT + ((size_t)b * NSP + n0) * CH, Bs);

  const int wv = threadIdx.x >> 5;
  const int o_base = (blockIdx.x * 4 + wv) * 32;
  const int l = lane_id();
  const int lc = l & 15, hv = l >> 4;

  v8f acc[2][4];
#pragma unroll
  for (int rt = 0; rt < 2; ++rt)
#pragma unroll
    for (int nt = 0; nt < 4; ++nt) acc[rt][nt] = zero_v8f();

#pragma unroll 2
  for (int kb = 0; kb < CH; kb += 32) {
    v16h a0 = load_a_frag(w, CH, o_base, kb);
    v16h a1 = load_a_frag(w, CH, o_base + 16, kb);
    __builtin_prefetch(w + (size_t)(o_base + lc) * CH + kb + 32, 0, 3);
#pragma unroll
    for (int nt = 0; nt < 4; ++nt) {
      v16h bf = load_b_frag(Bs, LDB, nt * 16, kb);
      acc[0][nt] = wmma_f16(a0, bf, acc[0][nt]);
      acc[1][nt] = wmma_f16(a1, bf, acc[1][nt]);
    }
  }

  const int seg = o_base >> 9;  // 0=q, 1=k, 2=v (uniform per wave)
#pragma unroll
  for (int rt = 0; rt < 2; ++rt) {
#pragma unroll
    for (int nt = 0; nt < 4; ++nt) {
      const int col = n0 + nt * 16 + lc;
#pragma unroll
      for (int v = 0; v < 8; ++v) {
        const int o = o_base + rt * 16 + v + 8 * hv;
        const float val = acc[rt][nt][v] + bias[o];
        const int oc = o & 511;
        const int hh = oc >> 6, c = oc & 63;
        const size_t bh = (size_t)(b * NHEADS + hh);
        if (seg == 0)      qT[(bh * NSP + col) * HD + c] = (_Float16)(val * 0.125f);
        else if (seg == 1) kT[(bh * NSP + col) * HD + c] = (_Float16)val;
        else               vbuf[(bh * HD + c) * NSP + col] = (_Float16)val;
      }
    }
  }
}

// ---------------- fused flash attention -> outT [B,N,C] f16 ----------------
__global__ void __launch_bounds__(128)
attention_kernel(const _Float16* __restrict__ qT, const _Float16* __restrict__ kT,
                 const _Float16* __restrict__ vbuf, _Float16* __restrict__ outT) {
  __shared__ _Float16 Plds[4][16 * PLD];
  const int wv = threadIdx.x >> 5;
  const int n_base = (blockIdx.x * 4 + wv) * 16;
  const int hh = blockIdx.y;
  const int b = blockIdx.z;
  const size_t bh = (size_t)(b * NHEADS + hh);
  const _Float16* Q = qT + bh * NSP * HD;   // [N, hd]
  const _Float16* K = kT + bh * NSP * HD;   // [N, hd]
  const _Float16* V = vbuf + bh * HD * NSP; // [hd, N]
  _Float16* Pw = &Plds[wv][0];

  const int l = lane_id();
  const int lc = l & 15, hv = l >> 4;

  float m_i[8], l_i[8];
  v8f o_acc[4];
#pragma unroll
  for (int v = 0; v < 8; ++v) { m_i[v] = -3.0e38f; l_i[v] = 0.f; }
#pragma unroll
  for (int t = 0; t < 4; ++t) o_acc[t] = zero_v8f();

  // Q fragments reused across all key blocks (softmax scale folded into q)
  const v16h aq0 = load_a_frag(Q, HD, n_base, 0);
  const v16h aq1 = load_a_frag(Q, HD, n_base, 32);

  for (int j = 0; j < NSP; j += 32) {
    v8f s0 = zero_v8f(), s1 = zero_v8f();
    s0 = wmma_f16(aq0, load_b_frag(K, HD, j, 0),       s0);
    s0 = wmma_f16(aq1, load_b_frag(K, HD, j, 32),      s0);
    s1 = wmma_f16(aq0, load_b_frag(K, HD, j + 16, 0),  s1);
    s1 = wmma_f16(aq1, load_b_frag(K, HD, j + 16, 32), s1);

#pragma unroll
    for (int v = 0; v < 8; ++v) {
      float t = fmaxf(s0[v], s1[v]);
      t = fmaxf(t, __shfl_xor(t, 1, 32));
      t = fmaxf(t, __shfl_xor(t, 2, 32));
      t = fmaxf(t, __shfl_xor(t, 4, 32));
      t = fmaxf(t, __shfl_xor(t, 8, 32));
      const float mnew = fmaxf(m_i[v], t);
      const float corr = __expf(m_i[v] - mnew);
      const float p0 = __expf(s0[v] - mnew);
      const float p1 = __expf(s1[v] - mnew);
      float rs = p0 + p1;
      rs += __shfl_xor(rs, 1, 32);
      rs += __shfl_xor(rs, 2, 32);
      rs += __shfl_xor(rs, 4, 32);
      rs += __shfl_xor(rs, 8, 32);
      l_i[v] = l_i[v] * corr + rs;
      m_i[v] = mnew;
#pragma unroll
      for (int t4 = 0; t4 < 4; ++t4) o_acc[t4][v] *= corr;
      // D-layout -> row-major P tile in LDS (per-wave region, padded stride)
      Pw[(v + 8 * hv) * PLD + lc]      = (_Float16)p0;
      Pw[(v + 8 * hv) * PLD + 16 + lc] = (_Float16)p1;
    }
    // O += P x V  (same-wave LDS write->read: DS pipe is in-order)
    const v16h ap = load_a_frag(Pw, PLD, 0, 0);
#pragma unroll
    for (int t4 = 0; t4 < 4; ++t4) {
      v16h bv = load_b_frag(V, NSP, t4 * 16, j);
      o_acc[t4] = wmma_f16(ap, bv, o_acc[t4]);
    }
  }
  // finalize: divide by row sum, write outT[b][n][h*64 + c]
  _Float16* outp = outT + (size_t)b * NSP * CH + (size_t)hh * HD;
#pragma unroll
  for (int v = 0; v < 8; ++v) {
    const int n = n_base + v + 8 * hv;
    const float inv = 1.0f / l_i[v];
#pragma unroll
    for (int t4 = 0; t4 < 4; ++t4) {
      outp[(size_t)n * CH + t4 * 16 + lc] = (_Float16)(o_acc[t4][v] * inv);
    }
  }
}

// -------- proj GEMM (LDS-staged B panel) + bias + residual (f32 out) --------
__global__ void __launch_bounds__(128)
proj_kernel(const _Float16* __restrict__ w, const float* __restrict__ bias,
            const _Float16* __restrict__ outT, const float* __restrict__ x,
            float* __restrict__ out) {
  __shared__ _Float16 Bs[64 * LDB];
  const int n0 = blockIdx.y * 64;
  const int b = blockIdx.z;
  stage_panel64(outT + ((size_t)b * NSP + n0) * CH, Bs);

  const int wv = threadIdx.x >> 5;
  const int o_base = (blockIdx.x * 4 + wv) * 32;
  const int l = lane_id();
  const int lc = l & 15, hv = l >> 4;

  v8f acc[2][4];
#pragma unroll
  for (int rt = 0; rt < 2; ++rt)
#pragma unroll
    for (int nt = 0; nt < 4; ++nt) acc[rt][nt] = zero_v8f();

#pragma unroll 2
  for (int kb = 0; kb < CH; kb += 32) {
    v16h a0 = load_a_frag(w, CH, o_base, kb);
    v16h a1 = load_a_frag(w, CH, o_base + 16, kb);
    __builtin_prefetch(w + (size_t)(o_base + lc) * CH + kb + 32, 0, 3);
#pragma unroll
    for (int nt = 0; nt < 4; ++nt) {
      v16h bf = load_b_frag(Bs, LDB, nt * 16, kb);
      acc[0][nt] = wmma_f16(a0, bf, acc[0][nt]);
      acc[1][nt] = wmma_f16(a1, bf, acc[1][nt]);
    }
  }

#pragma unroll
  for (int rt = 0; rt < 2; ++rt) {
#pragma unroll
    for (int nt = 0; nt < 4; ++nt) {
      const int col = n0 + nt * 16 + lc;
#pragma unroll
      for (int v = 0; v < 8; ++v) {
        const int o = o_base + rt * 16 + v + 8 * hv;
        const size_t idx = ((size_t)b * CH + o) * NSP + col;
        out[idx] = x[idx] + bias[o] + acc[rt][nt][v];
      }
    }
  }
}

extern "C" void kernel_launch(void* const* d_in, const int* in_sizes, int n_in,
                              void* d_out, int out_size, void* d_ws, size_t ws_size,
                              hipStream_t stream) {
  (void)in_sizes; (void)n_in; (void)out_size; (void)ws_size;
  const float* x      = (const float*)d_in[0];
  const float* norm_w = (const float*)d_in[1];
  const float* norm_b = (const float*)d_in[2];
  const float* qkv_w  = (const float*)d_in[3];
  const float* qkv_b  = (const float*)d_in[4];
  const float* proj_w = (const float*)d_in[5];
  const float* proj_b = (const float*)d_in[6];
  float* out = (float*)d_out;

  char* ws = (char*)d_ws;
  size_t off = 0;
  auto alloc = [&](size_t bytes) -> char* {
    char* p = ws + off;
    off = (off + bytes + 255) & ~(size_t)255;
    return p;
  };
  _Float16* xnT    = (_Float16*)alloc((size_t)BATCH * NSP * CH * 2);
  _Float16* qT     = (_Float16*)alloc((size_t)BATCH * NHEADS * NSP * HD * 2);
  _Float16* kT     = (_Float16*)alloc((size_t)BATCH * NHEADS * NSP * HD * 2);
  _Float16* vbuf   = (_Float16*)alloc((size_t)BATCH * NHEADS * HD * NSP * 2);
  _Float16* outT   = (_Float16*)alloc((size_t)BATCH * NSP * CH * 2);
  _Float16* qkvwh  = (_Float16*)alloc((size_t)QKV_O * CH * 2);
  _Float16* projwh = (_Float16*)alloc((size_t)CH * CH * 2);

  convert_weights_kernel<<<(QKV_O * CH + CH * CH + 255) / 256, 256, 0, stream>>>(
      qkv_w, proj_w, qkvwh, projwh);
  groupnorm_kernel<<<BATCH * NGRP, 256, 0, stream>>>(x, norm_w, norm_b, xnT);
  qkv_gemm_kernel<<<dim3(QKV_O / 128, NSP / 64, BATCH), 128, 0, stream>>>(
      qkvwh, qkv_b, xnT, qT, kT, vbuf);
  attention_kernel<<<dim3(NSP / 64, NHEADS, BATCH), 128, 0, stream>>>(
      qT, kT, vbuf, outT);
  proj_kernel<<<dim3(CH / 128, NSP / 64, BATCH), 128, 0, stream>>>(
      projwh, proj_b, outT, x, out);
}
